// EchoStateNetwork_5257039970688
// MI455X (gfx1250) — matrix operand order repaired and checked by reading, not compiled, not run on hardware
//
#include <hip/hip_runtime.h>

// Echo State Network on gfx1250 (MI455X), wave32 + WMMA f16.
//   Phase 0: init kernel zeros sync counters + state ping-pong (graph-replay safe).
//   Phase 1: inp[t,b,h] = u . W_in  via v_wmma_f32_16x16x32_f16.
//   Phase 2: persistent 64-WG scan; each WG owns 16 columns of H with its
//            1024x16 f16 W_res tile resident in LDS (32KB, loaded once);
//            per-step grid barrier via per-t atomic counters.

typedef _Float16 v16h __attribute__((ext_vector_type(16)));
typedef _Float16 v8h  __attribute__((ext_vector_type(8)));
typedef float    v8f  __attribute__((ext_vector_type(8)));
typedef float    v4f  __attribute__((ext_vector_type(4)));

#define LEAK   0.9f
#define HDIM   1024
#define BATCH  16
#define CHN    4
#define TLEN   1024
#define DDIM   64
#define FDIM   256          // CH*D
#define MROWS  64           // B*CH rows of state
#define NWG    64           // scan workgroups; each owns 16 columns of H

// ws layout (bytes)
#define WS_CNT_OFF    0           // u32 counters[TLEN]            (4 KB)
#define WS_STATE_OFF  8192        // f16 state[2][64][1024]        (256 KB)
#define WS_INP_OFF    524288      // f32 inp[T][B][H]              (64 MB)

__device__ __forceinline__ float fast_tanh(float x) {
    float ax = fabsf(x);
    float e  = __expf(-2.0f * ax);
    float r  = (1.0f - e) / (1.0f + e);
    return copysignf(r, x);
}

// ---------------------------------------------------------------- init ----
__global__ __launch_bounds__(256) void esn_init(unsigned* __restrict__ cnt,
                                                unsigned* __restrict__ state_words) {
    int i = blockIdx.x * blockDim.x + threadIdx.x;
    if (i < TLEN) cnt[i] = 0u;
    if (i < (2 * MROWS * HDIM) / 2) state_words[i] = 0u;   // zero both f16 buffers
}

// ------------------------------------------------------ input projection ----
// Grid: 16384 blocks x 128 threads; each wave computes one 16x16 tile of
// inp[t][b][h] (M-tile = all 16 batches at fixed t, N-tile = 16 h-columns).
__global__ __launch_bounds__(128) void esn_proj(const float* __restrict__ x,
                                                const float* __restrict__ Win,
                                                float* __restrict__ inp) {
    const int tid  = threadIdx.x;
    const int wave = tid >> 5, lane = tid & 31;
    const int tile = blockIdx.x * 4 + wave;        // 0..65535
    const int t    = tile >> 6;                    // 0..1023
    const int h0   = (tile & 63) << 4;
    const int ln   = lane & 15, kh = lane >> 4;

    v8f acc = {};
#pragma unroll
    for (int kk = 0; kk < 8; ++kk) {               // K = 256 in steps of 32
        // A fragment: u[t][b=ln][f]; layout: e[0..15]->K=kk*32+16*kh+0..15
        union { v16h v; _Float16 e[16]; } a;
#pragma unroll
        for (int chunk = 0; chunk < 2; ++chunk) {
            int f0 = kk * 32 + kh * 16 + chunk * 8;
            int c = f0 >> 6, d = f0 & 63;          // u[.,f] = x[b][c][t][d]
            const float* px = x + (((ln * CHN + c) * TLEN + t) * DDIM + d);
            v4f x0 = *(const v4f*)px;
            v4f x1 = *(const v4f*)(px + 4);
#pragma unroll
            for (int i = 0; i < 4; ++i) {
                a.e[chunk * 8 + i]     = (_Float16)x0[i];
                a.e[chunk * 8 + 4 + i] = (_Float16)x1[i];
            }
        }
        // B fragment: B[K][N] = Win[h0+ln][K]; lanes 0-15 -> K0=kk*32, 16-31 -> +16
        union { v16h v; _Float16 e[16]; } b;
        {
            int K0 = kk * 32 + kh * 16;
            const float* pw = Win + (h0 + ln) * FDIM + K0;
#pragma unroll
            for (int q = 0; q < 4; ++q) {
                v4f w = *(const v4f*)(pw + q * 4);
#pragma unroll
                for (int i = 0; i < 4; ++i) b.e[q * 4 + i] = (_Float16)w[i];
            }
        }
        acc = __builtin_amdgcn_wmma_f32_16x16x32_f16(false, a.v, false, b.v,
                                                     (short)0, acc, false, false);
    }
    // C layout: VGPR j -> M = j + 8*kh, N = ln
#pragma unroll
    for (int j = 0; j < 8; ++j) {
        int brow = j + kh * 8;
        inp[(t * BATCH + brow) * HDIM + h0 + ln] = acc[j];
    }
}

// ----------------------------------------------------------- scan (persistent) ----
__global__ __launch_bounds__(128) void esn_scan(const float* __restrict__ Wres,
                                                const float* __restrict__ inp,
                                                _Float16* __restrict__ stateBuf, // [2][64][1024]
                                                unsigned* __restrict__ cnt,      // [TLEN]
                                                float* __restrict__ out) {
    __shared__ _Float16 ldsB[32 * 32 * 16];        // 32 k-steps x 32 lanes x 16 halves = 32KB
    const int tid  = threadIdx.x;
    const int wave = tid >> 5, lane = tid & 31;
    const int ln   = lane & 15, kh = lane >> 4;
    const int n0   = blockIdx.x << 4;              // this WG's 16 columns of H

    // Pack this WG's W_res tile once, already in B-fragment lane layout:
    // idx = kk*512 + l*16 + i  -> value Wres[K][n0 + (l&15)], K = kk*32 + 16*(l>>4) + i
    for (int idx = tid; idx < 16384; idx += 128) {
        int i  = idx & 15;
        int l  = (idx >> 4) & 31;
        int kk = idx >> 9;
        int K  = kk * 32 + ((l >> 4) << 4) + i;
        ldsB[idx] = (_Float16)Wres[K * HDIM + n0 + (l & 15)];
    }
    __syncthreads();

    const int mA = wave * 16 + ln;                 // state row this lane loads for A

    for (int t = 0; t < TLEN; ++t) {
        const _Float16* Sprev = stateBuf + (t & 1) * (MROWS * HDIM);
        _Float16*       Snew  = stateBuf + ((t + 1) & 1) * (MROWS * HDIM);

        v8f acc0 = {}, acc1 = {};                  // 2 accumulators break WMMA RAW chain
#pragma unroll 4
        for (int kk = 0; kk < 32; ++kk) {
            union { v16h v; v8h h[2]; } a;
            const _Float16* pa = Sprev + mA * HDIM + kk * 32 + kh * 16;
            a.h[0] = *(const v8h*)pa;
            a.h[1] = *(const v8h*)(pa + 8);
            union { v16h v; v8h h[2]; } b;
            const _Float16* pb = &ldsB[kk * 512 + lane * 16];
            b.h[0] = *(const v8h*)pb;
            b.h[1] = *(const v8h*)(pb + 8);
            if (kk & 1)
                acc1 = __builtin_amdgcn_wmma_f32_16x16x32_f16(false, a.v, false, b.v,
                                                              (short)0, acc1, false, false);
            else
                acc0 = __builtin_amdgcn_wmma_f32_16x16x32_f16(false, a.v, false, b.v,
                                                              (short)0, acc0, false, false);
        }

        // Epilogue: new = leak*tanh(inp + upd) + (1-leak)*old
#pragma unroll
        for (int j = 0; j < 8; ++j) {
            int m = wave * 16 + j + kh * 8;        // state row (m = b*CH + c)
            int n = n0 + ln;                       // h index
            float upd  = acc0[j] + acc1[j];
            float ival = inp[(t * BATCH + (m >> 2)) * HDIM + n];
            float oldv = (float)Sprev[m * HDIM + n];
            float nv   = LEAK * fast_tanh(ival + upd) + (1.0f - LEAK) * oldv;
            Snew[m * HDIM + n] = (_Float16)nv;
            out[(m * TLEN + t) * HDIM + n] = nv;   // out[b][c][t][h]
        }

        // Grid barrier for step t: release new state, wait for all 64 WGs.
        __threadfence();
        __syncthreads();
        if (tid == 0) {
            __hip_atomic_fetch_add(&cnt[t], 1u, __ATOMIC_RELEASE, __HIP_MEMORY_SCOPE_AGENT);
            while (__hip_atomic_load(&cnt[t], __ATOMIC_ACQUIRE, __HIP_MEMORY_SCOPE_AGENT) < NWG) {
                __builtin_amdgcn_s_sleep(8);
            }
        }
        __syncthreads();
    }
}

// ---------------------------------------------------------------- launch ----
extern "C" void kernel_launch(void* const* d_in, const int* in_sizes, int n_in,
                              void* d_out, int out_size, void* d_ws, size_t ws_size,
                              hipStream_t stream) {
    const float* x    = (const float*)d_in[0];     // [16,4,1024,64]
    const float* Win  = (const float*)d_in[1];     // [1024,256]
    const float* Wres = (const float*)d_in[2];     // [1024,1024]
    float* out = (float*)d_out;                    // [16,4,1024,1024]

    char* ws = (char*)d_ws;
    unsigned* cnt      = (unsigned*)(ws + WS_CNT_OFF);
    _Float16* stateBuf = (_Float16*)(ws + WS_STATE_OFF);
    float*    inp      = (float*)(ws + WS_INP_OFF);

    esn_init<<<256, 256, 0, stream>>>(cnt, (unsigned*)stateBuf);
    esn_proj<<<(TLEN * (HDIM / 16)) / 4, 128, 0, stream>>>(x, Win, inp);
    esn_scan<<<NWG, 128, 0, stream>>>(Wres, inp, stateBuf, cnt, out);
}
